// DeformableTransformerDecoderAIGCV20_81973745811806
// MI455X (gfx1250) — compile-verified, matrix operand order checked by reading, not run on hardware
//
#include <hip/hip_runtime.h>
#include <hip/hip_bf16.h>
#include <math.h>

typedef __hip_bfloat16 bf16;
typedef __attribute__((ext_vector_type(16))) __bf16 v16bf;
typedef __attribute__((ext_vector_type(8)))  __bf16 v8bf;
typedef __attribute__((ext_vector_type(8)))  float  v8f;

#define BQ      2
#define NQ      1024
#define CDIM    256
#define HEADS   8
#define DHH     32
#define INNER   256
#define FFD     512
#define LEN_TOT 21504
#define NROWS   (BQ * NQ)          // 2048
#define SROWS   (BQ * LEN_TOT)     // 43008

// dynamic LDS for fused attention: xk (64x256 bf16) + kv (64x512 f32) + q (256 f32) + scores (8x64 f32)
#define ATTN_SMEM (32768 + 131072 + 1024 + 2048)

__device__ __forceinline__ void wait_async0() {
#if __has_builtin(__builtin_amdgcn_s_wait_asynccnt)
  __builtin_amdgcn_s_wait_asynccnt(0);
#else
  asm volatile("s_wait_asynccnt 0x0" ::: "memory");
#endif
}

// ---------------------------------------------------------------------------
// Fragment loader: 16-bit operand layout (ISA 7.12.2).
// Lane l<16 : K = {k0..k0+7} in elems 0..7, {k0+16..k0+23} in elems 8..15
// Lane l>=16: same +8.  p0 must already include the hi*8 offset.
// ---------------------------------------------------------------------------
__device__ __forceinline__ v16bf frag_ld(const bf16* p0) {
  v8bf lo = *reinterpret_cast<const v8bf*>(p0);
  v8bf hi = *reinterpret_cast<const v8bf*>(p0 + 16);
  v16bf r;
#pragma unroll
  for (int i = 0; i < 8; ++i) { r[i] = lo[i]; r[8 + i] = hi[i]; }
  return r;
}

// ---------------------------------------------------------------------------
// WMMA GEMM: out = act(A[MxK]bf16 @ B) (+bias)(+res); B given as Bt (Npad x K,
// zero-padded rows so B fragment loads need no guards).
// block = 128 threads = 4 waves; each wave owns a 16x32 output tile (2 wmma
// accumulators sharing one A fragment); block covers 16 rows x 128 cols.
// A panel (16 x K, contiguous in memory) is staged to LDS once per block via
// global_load_async_to_lds_b128.
// ---------------------------------------------------------------------------
__global__ void k_gemm(const bf16* __restrict__ A, const bf16* __restrict__ Bt,
                       const float* __restrict__ bias, const float* __restrict__ res,
                       float* __restrict__ outF, bf16* __restrict__ outB,
                       int M, int N, int Npad, int K, int act) {
  __shared__ bf16 As[16 * 512];                 // up to K=512
  const int t = threadIdx.x;
  const int row0 = blockIdx.y * 16;

  // ---- async stage of the contiguous 16xK A panel into LDS ----
  {
    const bf16* gA = A + (size_t)row0 * K;      // 16*K contiguous bf16
    const int chunks = (16 * K) >> 3;           // 16B chunks
    for (int ch = t; ch < chunks; ch += 128) {
      unsigned loff = (unsigned)(uintptr_t)As + (unsigned)(ch << 4);
      unsigned long long gaddr = (unsigned long long)(uintptr_t)(gA + (ch << 3));
      asm volatile("global_load_async_to_lds_b128 %0, %1, off"
                   :: "v"(loff), "v"(gaddr) : "memory");
    }
  }
  wait_async0();
  __syncthreads();

  const int lane = t & 31;
  const int wave = t >> 5;
  const int l  = lane & 15;
  const int hi = lane >> 4;
  const int col0 = blockIdx.x * 128 + wave * 32;
  if (col0 >= Npad) return;                     // wave-uniform skip

  const bf16* pA  = As + (size_t)l * K + hi * 8;                 // LDS
  const bf16* pB0 = Bt + (size_t)(col0 + l) * K + hi * 8;        // global
  const bf16* pB1 = Bt + (size_t)(col0 + 16 + l) * K + hi * 8;

  v8f acc0 = {}, acc1 = {};
  for (int k0 = 0; k0 < K; k0 += 32) {
    v16bf a  = frag_ld(pA + k0);
    v16bf b0 = frag_ld(pB0 + k0);
    v16bf b1 = frag_ld(pB1 + k0);
    __builtin_prefetch(pB0 + k0 + 64, 0, 1);    // global_prefetch_b8
    __builtin_prefetch(pB1 + k0 + 64, 0, 1);
    acc0 = __builtin_amdgcn_wmma_f32_16x16x32_bf16(false, a, false, b0,
                                                   (short)0, acc0, false, false);
    acc1 = __builtin_amdgcn_wmma_f32_16x16x32_bf16(false, a, false, b1,
                                                   (short)0, acc1, false, false);
  }

#pragma unroll
  for (int tile = 0; tile < 2; ++tile) {
    const int col = col0 + tile * 16 + l;
    const v8f& acc = tile ? acc1 : acc0;
    if (col < N) {
      const float bv = bias ? bias[col] : 0.0f;
#pragma unroll
      for (int v = 0; v < 8; ++v) {
        const int row = row0 + hi * 8 + v;
        const size_t idx = (size_t)row * N + col;
        float val = acc[v] + bv;
        if (act == 1) val = 0.5f * val * (1.0f + erff(val * 0.70710678118654752f));
        if (res)  val += res[idx];
        if (outF) outF[idx] = val;
        if (outB) outB[idx] = __float2bfloat16(val);
      }
    }
  }
}

// ---------------------------------------------------------------------------
// Weight transpose + bf16 convert + zero-pad:
//   W (KxN f32 row-major) -> Wt (Npad x K bf16), rows n>=N are zero.
// ---------------------------------------------------------------------------
__global__ void k_tr(const float* __restrict__ W, bf16* __restrict__ Wt,
                     int K, int N, int Npad) {
  int idx = blockIdx.x * 256 + threadIdx.x;
  if (idx >= Npad * K) return;
  int n = idx / K, k = idx % K;
  Wt[idx] = (n < N) ? __float2bfloat16(W[(size_t)k * N + n]) : __float2bfloat16(0.0f);
}

// ---------------------------------------------------------------------------
// cpe = center_pos[...,0,:] @ W_pos + b_pos
// ---------------------------------------------------------------------------
__global__ void k_cpe(const float* __restrict__ cpos, const float* __restrict__ Wp,
                      const float* __restrict__ bp, float* __restrict__ cpe) {
  int idx = blockIdx.x * 256 + threadIdx.x;    // NROWS*CDIM
  int row = idx >> 8, c = idx & 255;
  float p0 = cpos[(size_t)row * 6 + 0];
  float p1 = cpos[(size_t)row * 6 + 1];
  cpe[idx] = p0 * Wp[c] + p1 * Wp[CDIM + c] + bp[c];
}

// ---------------------------------------------------------------------------
// LayerNorm over C=256 (one row per block); optional pre-add / post-add;
// writes f32 and/or bf16.
// ---------------------------------------------------------------------------
__global__ void k_ln(const float* __restrict__ x, const float* __restrict__ addpre,
                     const float* __restrict__ addpost, const float* __restrict__ g,
                     const float* __restrict__ bb, float* __restrict__ outF,
                     bf16* __restrict__ outB) {
  __shared__ float red[256];
  const int row = blockIdx.x, t = threadIdx.x;
  const size_t idx = (size_t)row * CDIM + t;
  float v = x[idx] + (addpre ? addpre[idx] : 0.0f);
  red[t] = v; __syncthreads();
  for (int s = 128; s > 0; s >>= 1) { if (t < s) red[t] += red[t + s]; __syncthreads(); }
  const float mean = red[0] * (1.0f / CDIM);
  __syncthreads();
  const float d = v - mean;
  red[t] = d * d; __syncthreads();
  for (int s = 128; s > 0; s >>= 1) { if (t < s) red[t] += red[t + s]; __syncthreads(); }
  const float var = red[0] * (1.0f / CDIM);
  float y = d * rsqrtf(var + 1e-5f) * g[t] + bb[t];
  if (addpost) y += addpost[idx];
  if (outF) outF[idx] = y;
  if (outB) outB[idx] = __float2bfloat16(y);
}

// ---------------------------------------------------------------------------
// KNN: top-64 nearest (incl. self) per query via iterative argmin in LDS.
// Sorted ascending, so layer-0 (k=16) uses the first 16 indices.
// ---------------------------------------------------------------------------
__global__ void k_knn(const float* __restrict__ pos3d, int* __restrict__ knn_idx) {
  __shared__ float d2[NQ];
  __shared__ float rv[256];
  __shared__ int   ri[256];
  const int q = blockIdx.x, t = threadIdx.x;
  const int b = q / NQ, n = q % NQ;
  const float* pb = pos3d + (size_t)b * NQ * 3;
  const float px = pb[n * 3 + 0], py = pb[n * 3 + 1], pz = pb[n * 3 + 2];
  for (int j = t; j < NQ; j += 256) {
    float dx = pb[j * 3 + 0] - px, dy = pb[j * 3 + 1] - py, dz = pb[j * 3 + 2] - pz;
    d2[j] = dx * dx + dy * dy + dz * dz;
  }
  __syncthreads();
  for (int it = 0; it < 64; ++it) {
    float bv = 3.4e38f; int bi = 0x7fffffff;
    for (int j = t; j < NQ; j += 256) {
      float v = d2[j];
      if (v < bv || (v == bv && j < bi)) { bv = v; bi = j; }
    }
    rv[t] = bv; ri[t] = bi; __syncthreads();
    for (int s = 128; s > 0; s >>= 1) {
      if (t < s) {
        if (rv[t + s] < rv[t] || (rv[t + s] == rv[t] && ri[t + s] < ri[t])) {
          rv[t] = rv[t + s]; ri[t] = ri[t + s];
        }
      }
      __syncthreads();
    }
    if (t == 0) { knn_idx[(size_t)q * 64 + it] = ri[0]; d2[ri[0]] = 3.4e38f; }
    __syncthreads();
  }
}

// ---------------------------------------------------------------------------
// Fused KNN self-attention (one workgroup = one query, 8 waves):
//  1) gather k neighbor rows of h into LDS (bf16)
//  2) kv = xk @ Wkv via WMMA (A-fragments from LDS -> ds_load + v_wmma)
//  3) per-head softmax attention in VALU, bf16 output
// ---------------------------------------------------------------------------
__global__ void k_attn(const float* __restrict__ hsrc, const float* __restrict__ qsa,
                       const int* __restrict__ knn_idx, const bf16* __restrict__ WkvT,
                       bf16* __restrict__ saOut, int kNb) {
  extern __shared__ char smem[];
  bf16*  xk = (bf16*)smem;                               // 64 x 256
  float* kv = (float*)(smem + 32768);                    // 64 x 512
  float* qv = (float*)(smem + 32768 + 131072);           // 256
  float* sc = (float*)(smem + 32768 + 131072 + 1024);    // 8 x 64

  const int qid = blockIdx.x;
  const int b = qid / NQ;
  const int t = threadIdx.x;
  const int* kid = knn_idx + (size_t)qid * 64;

  for (int e = t; e < kNb * CDIM; e += 256) {
    const int s = e >> 8, c = e & 255;
    const int row = kid[s];
    xk[s * CDIM + c] = __float2bfloat16(hsrc[((size_t)b * NQ + row) * CDIM + c]);
  }
  if (t < CDIM) qv[t] = qsa[(size_t)qid * CDIM + t];
  __syncthreads();

  const int lane = t & 31, wave = t >> 5;
  const int l = lane & 15, hi = lane >> 4;
  const int ntr = kNb >> 4;                  // 1 or 4 row tiles
  const int ntiles = ntr * 32;               // x 32 column tiles (512 cols)
  for (int tile = wave; tile < ntiles; tile += 8) {
    const int tr = tile >> 5, tc = tile & 31;
    v8f acc = {};
    const bf16* pA = xk + (size_t)(tr * 16 + l) * CDIM + hi * 8;
    const bf16* pB = WkvT + (size_t)(tc * 16 + l) * CDIM + hi * 8;
    for (int k0 = 0; k0 < CDIM; k0 += 32) {
      v16bf a = frag_ld(pA + k0);
      v16bf bm = frag_ld(pB + k0);
      acc = __builtin_amdgcn_wmma_f32_16x16x32_bf16(false, a, false, bm,
                                                    (short)0, acc, false, false);
    }
    const int col = tc * 16 + l;
#pragma unroll
    for (int v = 0; v < 8; ++v) {
      const int s = tr * 16 + hi * 8 + v;
      kv[s * 512 + col] = acc[v];
    }
  }
  __syncthreads();

  // scores: wave h handles head h
  {
    const int h = wave;
    for (int s = lane; s < kNb; s += 32) {
      float dot = 0.0f;
#pragma unroll
      for (int d = 0; d < DHH; ++d) dot += qv[h * DHH + d] * kv[s * 512 + h * DHH + d];
      sc[h * 64 + s] = dot * 0.17677669529663687f;   // 1/sqrt(32)
    }
  }
  __syncthreads();
  if (t < HEADS) {
    float m = -3.4e38f;
    for (int s = 0; s < kNb; ++s) m = fmaxf(m, sc[t * 64 + s]);
    float sum = 0.0f;
    for (int s = 0; s < kNb; ++s) { float e = __expf(sc[t * 64 + s] - m); sc[t * 64 + s] = e; sum += e; }
    const float inv = 1.0f / sum;
    for (int s = 0; s < kNb; ++s) sc[t * 64 + s] *= inv;
  }
  __syncthreads();
  {
    const int h = t >> 5, d = t & 31;
    float acc = 0.0f;
    for (int s = 0; s < kNb; ++s)
      acc += sc[h * 64 + s] * kv[s * 512 + INNER + h * DHH + d];
    saOut[(size_t)qid * INNER + h * DHH + d] = __float2bfloat16(acc);
  }
}

// ---------------------------------------------------------------------------
// MS-deformable sampling: one workgroup per query; stage per-(h,l,p) locations
// and softmaxed attention weights in LDS; thread (h,d) accumulates 27 bilinear
// samples from `value`.
// ---------------------------------------------------------------------------
__global__ void k_msdeform(const float* __restrict__ offb, const float* __restrict__ awb,
                           const float* __restrict__ cpos, const float* __restrict__ value,
                           bf16* __restrict__ msOut) {
  __shared__ float lx[8 * 27], ly[8 * 27], lw[8 * 27];
  const int Hs[3] = {128, 64, 32};
  const int Ws[3] = {128, 64, 32};
  const int st[3] = {0, 16384, 20480};
  const int qid = blockIdx.x, t = threadIdx.x;
  const int b = qid / NQ, n = qid % NQ;

  if (t < 216) {
    const int h = t / 27, j = t % 27, lvl = j / 9, p = j % 9;
    const float ox = offb[(size_t)qid * 432 + h * 54 + lvl * 18 + p * 2 + 0];
    const float oy = offb[(size_t)qid * 432 + h * 54 + lvl * 18 + p * 2 + 1];
    const float rx = cpos[((size_t)(b * NQ + n) * 3 + lvl) * 2 + 0];
    const float ry = cpos[((size_t)(b * NQ + n) * 3 + lvl) * 2 + 1];
    lx[t] = rx + ox / (float)Ws[lvl];
    ly[t] = ry + oy / (float)Hs[lvl];
    lw[t] = awb[(size_t)qid * 216 + h * 27 + j];
  }
  __syncthreads();
  if (t < HEADS) {
    float m = -3.4e38f;
    for (int j = 0; j < 27; ++j) m = fmaxf(m, lw[t * 27 + j]);
    float sum = 0.0f;
    for (int j = 0; j < 27; ++j) { float e = __expf(lw[t * 27 + j] - m); lw[t * 27 + j] = e; sum += e; }
    const float inv = 1.0f / sum;
    for (int j = 0; j < 27; ++j) lw[t * 27 + j] *= inv;
  }
  __syncthreads();

  const int h = t >> 5, d = t & 31;
  float acc = 0.0f;
  for (int lvl = 0; lvl < 3; ++lvl) {
    const int Hl = Hs[lvl], Wl = Ws[lvl], base = st[lvl];
    for (int p = 0; p < 9; ++p) {
      const int j = h * 27 + lvl * 9 + p;
      const float xf = lx[j] * (float)Wl - 0.5f;
      const float yf = ly[j] * (float)Hl - 0.5f;
      const float w = lw[j];
      const float x0 = floorf(xf), y0 = floorf(yf);
#pragma unroll
      for (int dy = 0; dy < 2; ++dy)
#pragma unroll
        for (int dx = 0; dx < 2; ++dx) {
          const float ix = x0 + dx, iy = y0 + dy;
          const float bw = (1.0f - fabsf(xf - ix)) * (1.0f - fabsf(yf - iy));
          if (ix >= 0.0f && ix <= (float)(Wl - 1) && iy >= 0.0f && iy <= (float)(Hl - 1)) {
            const int xi = (int)ix, yi = (int)iy;
            acc += w * bw * value[((size_t)b * LEN_TOT + base + yi * Wl + xi) * INNER + h * DHH + d];
          }
        }
    }
  }
  msOut[(size_t)qid * INNER + h * DHH + d] = __float2bfloat16(acc);
}

// ---------------------------------------------------------------------------
extern "C" void kernel_launch(void* const* d_in, const int* in_sizes, int n_in,
                              void* d_out, int out_size, void* d_ws, size_t ws_size,
                              hipStream_t stream) {
  (void)in_sizes; (void)n_in; (void)out_size; (void)ws_size;

  const float* x     = (const float*)d_in[0];
  const float* src   = (const float*)d_in[1];
  const float* cpos  = (const float*)d_in[2];
  const float* pos3  = (const float*)d_in[3];
  const float* Wpos  = (const float*)d_in[6];
  const float* bpos  = (const float*)d_in[7];
  const float* ln1g  = (const float*)d_in[8];
  const float* ln1b  = (const float*)d_in[9];
  const float* ln2g  = (const float*)d_in[10];
  const float* ln2b  = (const float*)d_in[11];
  const float* ln3g  = (const float*)d_in[12];
  const float* ln3b  = (const float*)d_in[13];
  const float* Wq    = (const float*)d_in[14];
  const float* Wkv   = (const float*)d_in[15];
  const float* Wosa  = (const float*)d_in[16];
  const float* bosa  = (const float*)d_in[17];
  const float* Wval  = (const float*)d_in[18];
  const float* bval  = (const float*)d_in[19];
  const float* Woff  = (const float*)d_in[20];
  const float* boff  = (const float*)d_in[21];
  const float* Waw   = (const float*)d_in[22];
  const float* baw   = (const float*)d_in[23];
  const float* Woca  = (const float*)d_in[24];
  const float* boca  = (const float*)d_in[25];
  const float* W1    = (const float*)d_in[26];
  const float* b1    = (const float*)d_in[27];
  const float* W2    = (const float*)d_in[28];
  const float* b2    = (const float*)d_in[29];

  // padded column counts (multiple of 32) for transposed weights
  const int NP_OFF = 448;   // 432 logical
  const int NP_AW  = 224;   // 216 logical

  // ---- workspace carve ----
  char* ws = (char*)d_ws; size_t pos = 0;
  auto carve = [&](size_t bytes) -> void* {
    void* p = ws + pos; pos = (pos + bytes + 255) & ~(size_t)255; return p;
  };
  float* xbuf   = (float*)carve((size_t)NROWS * CDIM * 4);
  float* cpe    = (float*)carve((size_t)NROWS * CDIM * 4);
  float* hbuf   = (float*)carve((size_t)NROWS * CDIM * 4);
  bf16*  hbf    = (bf16*) carve((size_t)NROWS * CDIM * 2);
  float* qsa    = (float*)carve((size_t)NROWS * INNER * 4);
  int*   knnidx = (int*)  carve((size_t)NROWS * 64 * 4);
  bf16*  sabf   = (bf16*) carve((size_t)NROWS * INNER * 2);
  bf16*  qmsbf  = (bf16*) carve((size_t)NROWS * CDIM * 2);
  bf16*  srcbf  = (bf16*) carve((size_t)SROWS * CDIM * 2);
  float* value  = (float*)carve((size_t)SROWS * INNER * 4);
  float* offb   = (float*)carve((size_t)NROWS * 432 * 4);
  float* awb    = (float*)carve((size_t)NROWS * 216 * 4);
  bf16*  msbf   = (bf16*) carve((size_t)NROWS * INNER * 2);
  bf16*  ffbf   = (bf16*) carve((size_t)NROWS * FFD * 2);
  bf16*  WqT    = (bf16*) carve((size_t)2 * CDIM * INNER * 2);
  bf16*  WkvT   = (bf16*) carve((size_t)2 * CDIM * 2 * INNER * 2);
  bf16*  WosaT  = (bf16*) carve((size_t)2 * INNER * CDIM * 2);
  bf16*  WvalT  = (bf16*) carve((size_t)2 * CDIM * INNER * 2);
  bf16*  WoffT  = (bf16*) carve((size_t)2 * CDIM * NP_OFF * 2);
  bf16*  WawT   = (bf16*) carve((size_t)2 * CDIM * NP_AW * 2);
  bf16*  WocaT  = (bf16*) carve((size_t)2 * INNER * CDIM * 2);
  bf16*  W1T    = (bf16*) carve((size_t)2 * CDIM * FFD * 2);
  bf16*  W2T    = (bf16*) carve((size_t)2 * FFD * CDIM * 2);

  hipFuncSetAttribute((const void*)k_attn,
                      hipFuncAttributeMaxDynamicSharedMemorySize, ATTN_SMEM);

  hipMemcpyAsync(xbuf, x, (size_t)NROWS * CDIM * 4, hipMemcpyDeviceToDevice, stream);

  auto tr = [&](const float* W, bf16* Wt, int K, int N, int Npad) {
    int tot = K * Npad;
    k_tr<<<(tot + 255) / 256, 256, 0, stream>>>(W, Wt, K, N, Npad);
  };
  for (int i = 0; i < 2; ++i) {
    tr(Wq   + (size_t)i * CDIM * INNER,     WqT   + (size_t)i * CDIM * INNER,     CDIM, INNER, INNER);
    tr(Wkv  + (size_t)i * CDIM * 2 * INNER, WkvT  + (size_t)i * CDIM * 2 * INNER, CDIM, 2 * INNER, 2 * INNER);
    tr(Wosa + (size_t)i * INNER * CDIM,     WosaT + (size_t)i * INNER * CDIM,     INNER, CDIM, CDIM);
    tr(Wval + (size_t)i * CDIM * INNER,     WvalT + (size_t)i * CDIM * INNER,     CDIM, INNER, INNER);
    tr(Woff + (size_t)i * CDIM * 432,       WoffT + (size_t)i * CDIM * NP_OFF,    CDIM, 432, NP_OFF);
    tr(Waw  + (size_t)i * CDIM * 216,       WawT  + (size_t)i * CDIM * NP_AW,     CDIM, 216, NP_AW);
    tr(Woca + (size_t)i * INNER * CDIM,     WocaT + (size_t)i * INNER * CDIM,     INNER, CDIM, CDIM);
    tr(W1   + (size_t)i * CDIM * FFD,       W1T   + (size_t)i * CDIM * FFD,       CDIM, FFD, FFD);
    tr(W2   + (size_t)i * FFD * CDIM,       W2T   + (size_t)i * FFD * CDIM,       FFD, CDIM, CDIM);
  }

  k_cpe<<<NROWS, 256, 0, stream>>>(cpos, Wpos, bpos, cpe);
  k_knn<<<NROWS, 256, 0, stream>>>(pos3, knnidx);

  auto gemm = [&](const bf16* A, const bf16* Bt, const float* bias, const float* res,
                  float* oF, bf16* oB, int M, int N, int Npad, int K, int act) {
    dim3 g((Npad + 127) / 128, M / 16);
    k_gemm<<<g, 128, 0, stream>>>(A, Bt, bias, res, oF, oB, M, N, Npad, K, act);
  };

  for (int i = 0; i < 2; ++i) {
    const bf16* wqT   = WqT   + (size_t)i * CDIM * INNER;
    const bf16* wkvT  = WkvT  + (size_t)i * CDIM * 2 * INNER;
    const bf16* wosaT = WosaT + (size_t)i * INNER * CDIM;
    const bf16* wvalT = WvalT + (size_t)i * CDIM * INNER;
    const bf16* woffT = WoffT + (size_t)i * CDIM * NP_OFF;
    const bf16* wawT  = WawT  + (size_t)i * CDIM * NP_AW;
    const bf16* wocaT = WocaT + (size_t)i * INNER * CDIM;
    const bf16* w1T   = W1T   + (size_t)i * CDIM * FFD;
    const bf16* w2T   = W2T   + (size_t)i * FFD * CDIM;

    // h = LN(x + cpe)
    k_ln<<<NROWS, 256, 0, stream>>>(xbuf, cpe, nullptr, ln1g + i * CDIM, ln1b + i * CDIM, hbuf, hbf);
    // q = h @ Wq
    gemm(hbf, wqT, nullptr, nullptr, qsa, nullptr, NROWS, INNER, INNER, CDIM, 0);
    // fused gather + kv-projection + softmax attention
    k_attn<<<NROWS, 256, ATTN_SMEM, stream>>>(hbuf, qsa, knnidx, wkvT, sabf, (i == 0) ? 16 : 64);
    // x += sa @ Wo_sa + bo_sa
    gemm(sabf, wosaT, bosa + i * CDIM, xbuf, xbuf, nullptr, NROWS, CDIM, CDIM, INNER, 0);

    // q_ms = LN(x) + cpe ; s = LN(src)
    k_ln<<<NROWS, 256, 0, stream>>>(xbuf, nullptr, cpe, ln2g + i * CDIM, ln2b + i * CDIM, nullptr, qmsbf);
    k_ln<<<SROWS, 256, 0, stream>>>(src, nullptr, nullptr, ln2g + i * CDIM, ln2b + i * CDIM, nullptr, srcbf);
    // value = s @ Wval + bval
    gemm(srcbf, wvalT, bval + i * INNER, nullptr, value, nullptr, SROWS, INNER, INNER, CDIM, 0);
    // off / aw
    gemm(qmsbf, woffT, boff + i * 432, nullptr, offb, nullptr, NROWS, 432, NP_OFF, CDIM, 0);
    gemm(qmsbf, wawT,  baw  + i * 216, nullptr, awb,  nullptr, NROWS, 216, NP_AW, CDIM, 0);
    // bilinear deformable sampling
    k_msdeform<<<NROWS, 256, 0, stream>>>(offb, awb, cpos, value, msbf);
    // x += ms @ Wo_ca + bo_ca
    gemm(msbf, wocaT, boca + i * CDIM, xbuf, xbuf, nullptr, NROWS, CDIM, CDIM, INNER, 0);

    // FFN
    k_ln<<<NROWS, 256, 0, stream>>>(xbuf, nullptr, nullptr, ln3g + i * CDIM, ln3b + i * CDIM, nullptr, hbf);
    gemm(hbf, w1T, b1 + i * FFD, nullptr, nullptr, ffbf, NROWS, FFD, FFD, CDIM, 1);
    gemm(ffbf, w2T, b2 + i * CDIM, xbuf, xbuf, nullptr, NROWS, CDIM, CDIM, FFD, 0);
  }

  hipMemcpyAsync(d_out, xbuf, (size_t)NROWS * CDIM * 4, hipMemcpyDeviceToDevice, stream);
}
